// PDRiskNetActorCritic_88184268521856
// MI455X (gfx1250) — matrix operand order repaired and checked by reading, not compiled, not run on hardware
//
#include <hip/hip_runtime.h>
#include <hip/hip_bf16.h>
#include <math.h>

typedef __attribute__((ext_vector_type(16))) _Float16 v16h;
typedef __attribute__((ext_vector_type(8)))  float    v8f;

#define B_        256
#define T_        10
#define N_        1024
#define PROPRIO_  48
#define OBS_STRIDE_ (PROPRIO_ + T_*N_*3)   // 30768
#define ENC_      64
#define HP_       187
#define HD_       64
#define LAT_PAD_  320    // 299 padded to multiple of 32

// Branch-free activations built on the hardware transcendental (v_exp_f32).
__device__ __forceinline__ float eluf(float x) {
    return fmaxf(x, 0.0f) + (__expf(fminf(x, 0.0f)) - 1.0f);
}
__device__ __forceinline__ float sigmf(float x) {
    return 1.0f / (1.0f + __expf(-x));
}
__device__ __forceinline__ float tanh_fast(float x) {
    return 1.0f - 2.0f / (__expf(2.0f * x) + 1.0f);
}

// ---------------------------------------------------------------------------
// 1) Distance-bin means: bin_ids[i] = i*dk//m is non-decreasing, so bin k owns
//    indices [ceil(k*m/dk), ceil((k+1)*m/dk)). One thread per bin.
// ---------------------------------------------------------------------------
__global__ __launch_bounds__(256) void bin_dist_kernel(
    const float* __restrict__ obs, const int* __restrict__ dist_idx,
    const float* __restrict__ counts, int m, int dk,
    float* __restrict__ dist_pts)
{
    const int bt = blockIdx.x;
    const int b = bt / T_, t = bt % T_;
    const float* lidar = obs + (size_t)b * OBS_STRIDE_ + PROPRIO_ + (size_t)t * N_ * 3;
    for (int k = threadIdx.x; k < dk; k += 256) {
        const int start = (k * m + dk - 1) / dk;
        const int end   = ((k + 1) * m + dk - 1) / dk;
        float sx = 0.f, sy = 0.f, sz = 0.f;
        for (int i = start; i < end; ++i) {
            const float* p = lidar + (size_t)dist_idx[i] * 3;
            sx += p[0]; sy += p[1]; sz += p[2];
        }
        const float inv = 1.0f / fmaxf(counts[k], 1.0f);
        float* o = dist_pts + ((size_t)bt * dk + k) * 3;
        o[0] = sx * inv; o[1] = sy * inv; o[2] = sz * inv;
    }
}

// ---------------------------------------------------------------------------
// 2) Point encoder + mean pool. One block per (b,t). Layer1 (3->64, elu) in
//    VALU to f16 LDS; layer2 (64->64) via v_wmma_f32_16x16x32_f16; branchless
//    elu + masked mean-pool via per-lane partials + LDS float atomics.
//    Sorting in the reference is a no-op under mean pooling -> skipped.
// ---------------------------------------------------------------------------
__global__ __launch_bounds__(256) void enc_pool_kernel(
    const float* __restrict__ obs,
    const int*   __restrict__ gather_idx,   // non-null: prox branch (gather from lidar)
    const float* __restrict__ pts,          // non-null: dist branch (binned points)
    int P,
    const float* __restrict__ W1, const float* __restrict__ b1,
    const float* __restrict__ W2, const float* __restrict__ b2,
    float* __restrict__ seq_out)
{
    __shared__ _Float16 sH1[128][ENC_];     // 16 KB chunk of layer-1 activations
    __shared__ _Float16 sW2[ENC_ * ENC_];   // 8 KB
    __shared__ float sW1[ENC_ * 3];
    __shared__ float sB1[ENC_];
    __shared__ float sB2[ENC_];
    __shared__ float sAcc[ENC_];

    const int tid = threadIdx.x;
    const int bt  = blockIdx.x;
    const int b   = bt / T_;
    const int t   = bt % T_;
    const float* lidar = obs + (size_t)b * OBS_STRIDE_ + PROPRIO_ + (size_t)t * N_ * 3;

    for (int i = tid; i < ENC_ * ENC_; i += 256) sW2[i] = (_Float16)W2[i];
    for (int i = tid; i < ENC_ * 3;   i += 256) sW1[i] = W1[i];
    if (tid < ENC_) { sB1[tid] = b1[tid]; sB2[tid] = b2[tid]; sAcc[tid] = 0.0f; }
    __syncthreads();

    const int lane = tid & 31;
    const int wv   = tid >> 5;         // wave id 0..7 -> M-tile within chunk
    const int col  = lane & 15;
    const int hi   = (lane >> 4) & 1;

    // Preload W2 B-fragments (B[k][n] = W2[n][k], row n contiguous in K)
    v16h Bf[4][2];
    #pragma unroll
    for (int nt = 0; nt < 4; ++nt)
        #pragma unroll
        for (int ks = 0; ks < 2; ++ks) {
            const _Float16* src = &sW2[(nt * 16 + col) * ENC_ + ks * 32 + hi * 16];
            #pragma unroll
            for (int j = 0; j < 16; ++j) Bf[nt][ks][j] = src[j];
        }

    float acc[4] = {0.f, 0.f, 0.f, 0.f};
    const int nchunks = (P + 127) >> 7;
    for (int c = 0; c < nchunks; ++c) {
        // ---- layer 1: 2 threads per point, 32 channels each (branchless) ----
        const int ptl  = tid >> 1;
        const int half = tid & 1;
        const int p    = c * 128 + ptl;
        float x = 0.f, y = 0.f, z = 0.f;
        const float vmask = (p < P) ? 1.0f : 0.0f;
        if (p < P) {  // guard only the OOB global read
            const float* src = gather_idx ? (lidar + (size_t)gather_idx[p] * 3)
                                          : (pts + ((size_t)bt * P + p) * 3);
            x = src[0]; y = src[1]; z = src[2];
        }
        #pragma unroll
        for (int oo = 0; oo < 32; ++oo) {
            const int o = half * 32 + oo;
            float h = fmaf(sW1[o * 3 + 0], x, fmaf(sW1[o * 3 + 1], y,
                      fmaf(sW1[o * 3 + 2], z, sB1[o])));
            h = vmask * eluf(h);
            sH1[ptl][o] = (_Float16)h;
        }
        __syncthreads();

        // ---- layer 2: WMMA over K=64 (two 16x16x32 steps), 4 N-tiles ----
        const _Float16* arow = &sH1[wv * 16 + col][0];
        const int ko = hi * 16;
        v16h A0, A1;
        #pragma unroll
        for (int j = 0; j < 8; ++j) {
            A0[j]     = arow[ko + j];
            A0[8 + j] = arow[8 + ko + j];
            A1[j]     = arow[32 + ko + j];
            A1[8 + j] = arow[40 + ko + j];
        }
        const int rbase = c * 128 + wv * 16 + hi * 8;
        #pragma unroll
        for (int nt = 0; nt < 4; ++nt) {
            v8f cf = {};
            cf = __builtin_amdgcn_wmma_f32_16x16x32_f16(false, A0, false, Bf[nt][0],
                                                        (short)0, cf, false, false);
            cf = __builtin_amdgcn_wmma_f32_16x16x32_f16(false, A1, false, Bf[nt][1],
                                                        (short)0, cf, false, false);
            const float bias = sB2[nt * 16 + col];
            float s = 0.f;
            #pragma unroll
            for (int v = 0; v < 8; ++v) {
                const float msk = (rbase + v < P) ? 1.0f : 0.0f;   // branchless row mask
                s = fmaf(msk, eluf(cf[v] + bias), s);
            }
            acc[nt] += s;
        }
        __syncthreads();
    }
    #pragma unroll
    for (int nt = 0; nt < 4; ++nt) atomicAdd(&sAcc[nt * 16 + col], acc[nt]);
    __syncthreads();
    if (tid < ENC_) seq_out[(size_t)bt * ENC_ + tid] = sAcc[tid] / (float)P;
}

// ---------------------------------------------------------------------------
// 3) One GRU time step, one block per batch row; x,h staged in LDS, weight
//    rows stream through L2 (shared across all 256 blocks).
// ---------------------------------------------------------------------------
__global__ __launch_bounds__(256) void gru_step_kernel(
    const float* __restrict__ xseq, int t,
    const float* __restrict__ Wih, const float* __restrict__ Whh,
    const float* __restrict__ bih, const float* __restrict__ bhh,
    const float* __restrict__ hin, float* __restrict__ hout, int H)
{
    __shared__ float sx[ENC_];
    __shared__ float sh[192];
    const int b = blockIdx.x;
    const int tid = threadIdx.x;
    if (tid < ENC_) sx[tid] = xseq[((size_t)b * T_ + t) * ENC_ + tid];
    if (tid < H)    sh[tid] = hin[(size_t)b * H + tid];
    __syncthreads();
    for (int j = tid; j < H; j += 256) {
        const float* wr = Wih + (size_t)j * ENC_;
        const float* wz = Wih + (size_t)(H + j) * ENC_;
        const float* wn = Wih + (size_t)(2 * H + j) * ENC_;
        float ir = bih[j], iz = bih[H + j], inn = bih[2 * H + j];
        #pragma unroll 4
        for (int k = 0; k < ENC_; ++k) {
            const float xv = sx[k];
            ir = fmaf(wr[k], xv, ir); iz = fmaf(wz[k], xv, iz); inn = fmaf(wn[k], xv, inn);
        }
        const float* ur = Whh + (size_t)j * H;
        const float* uz = Whh + (size_t)(H + j) * H;
        const float* un = Whh + (size_t)(2 * H + j) * H;
        float hr = bhh[j], hz = bhh[H + j], hn = bhh[2 * H + j];
        #pragma unroll 4
        for (int k = 0; k < H; ++k) {
            const float hv = sh[k];
            hr = fmaf(ur[k], hv, hr); hz = fmaf(uz[k], hv, hz); hn = fmaf(un[k], hv, hn);
        }
        const float r = sigmf(ir + hr);
        const float z = sigmf(iz + hz);
        const float n = tanh_fast(inn + r * hn);
        hout[(size_t)b * H + j] = (1.f - z) * n + z * sh[j];
    }
}

__global__ void zero_f32_kernel(float* p, int n) {
    const int i = blockIdx.x * blockDim.x + threadIdx.x;
    if (i < n) p[i] = 0.f;
}

// ---------------------------------------------------------------------------
// 4) latent = [proprio | h_prox | h_dist] -> f16, zero-padded to 320
// ---------------------------------------------------------------------------
__global__ void build_latent_kernel(const float* __restrict__ obs,
                                    const float* __restrict__ hp,
                                    const float* __restrict__ hd,
                                    _Float16* __restrict__ lat)
{
    const int b = blockIdx.x;
    for (int j = threadIdx.x; j < LAT_PAD_; j += blockDim.x) {
        float v = 0.f;
        if (j < PROPRIO_)                  v = obs[(size_t)b * OBS_STRIDE_ + j];
        else if (j < PROPRIO_ + HP_)       v = hp[(size_t)b * HP_ + (j - PROPRIO_)];
        else if (j < PROPRIO_ + HP_ + HD_) v = hd[(size_t)b * HD_ + (j - PROPRIO_ - HP_)];
        lat[(size_t)b * LAT_PAD_ + j] = (_Float16)v;
    }
}

// ---------------------------------------------------------------------------
// 5) Generic FC layer via WMMA: one wave computes a 16x16 output tile,
//    looping K in steps of 32. A is f16 (zero-padded to KPAD); W,bias are f32.
//    act_elu: elu -> f16 (zero-padded to NPAD); else linear -> f32 (final).
// ---------------------------------------------------------------------------
__global__ __launch_bounds__(32) void fc_wmma_kernel(
    const _Float16* __restrict__ A, int KPAD, int K,
    const float* __restrict__ W, const float* __restrict__ bias,
    int N, int NPAD, int act_elu,
    _Float16* __restrict__ outH, float* __restrict__ outF)
{
    const int mt   = blockIdx.x;
    const int nt   = blockIdx.y;
    const int lane = threadIdx.x & 31;
    const int col  = lane & 15;
    const int hi   = (lane >> 4) & 1;
    const int arow = mt * 16 + col;
    const int ncol = nt * 16 + col;
    const _Float16* abase = A + (size_t)arow * KPAD;
    const float*    wrow  = W + (size_t)ncol * K;
    v8f c = {};
    for (int k0 = 0; k0 < KPAD; k0 += 32) {
        v16h a, bf;
        const int ko = hi * 16;
        #pragma unroll
        for (int j = 0; j < 8; ++j) {
            a[j]     = abase[k0 + ko + j];
            a[8 + j] = abase[k0 + 8 + ko + j];
        }
        const int kb = k0 + hi * 16;
        if (ncol < N && kb + 16 <= K) {
            // fast path: contiguous, unguarded -> vectorizable global loads
            #pragma unroll
            for (int j = 0; j < 16; ++j) bf[j] = (_Float16)wrow[kb + j];
        } else {
            #pragma unroll
            for (int j = 0; j < 16; ++j) {
                const int kk = kb + j;
                bf[j] = (_Float16)((ncol < N && kk < K) ? wrow[kk] : 0.f);
            }
        }
        c = __builtin_amdgcn_wmma_f32_16x16x32_f16(false, a, false, bf,
                                                   (short)0, c, false, false);
    }
    const float bv = (ncol < N) ? bias[ncol] : 0.f;
    const float nmask = (ncol < N) ? 1.0f : 0.0f;
    const int rbase = mt * 16 + hi * 8;
    if (act_elu) {
        #pragma unroll
        for (int v = 0; v < 8; ++v) {
            const float yv = nmask * eluf(c[v] + bv);   // branchless; pad cols -> 0
            outH[(size_t)(rbase + v) * NPAD + ncol] = (_Float16)yv;
        }
    } else if (ncol < N) {
        #pragma unroll
        for (int v = 0; v < 8; ++v)
            outF[(size_t)(rbase + v) * N + ncol] = c[v] + bv;
    }
}

// ---------------------------------------------------------------------------
extern "C" void kernel_launch(void* const* d_in, const int* in_sizes, int n_in,
                              void* d_out, int out_size, void* d_ws, size_t ws_size,
                              hipStream_t stream)
{
    (void)n_in; (void)out_size; (void)ws_size;
    const float* obs     = (const float*)d_in[0];
    const int*   proxIdx = (const int*)d_in[1];
    const int*   distIdx = (const int*)d_in[2];
    /* d_in[3] = bin_ids: unused, bins derived arithmetically */
    const float* counts  = (const float*)d_in[4];
    const int pk = in_sizes[1];
    const int m  = in_sizes[2];
    const int dk = in_sizes[4];

    const float* peP_W1 = (const float*)d_in[5];
    const float* peP_b1 = (const float*)d_in[6];
    const float* peP_W2 = (const float*)d_in[7];
    const float* peP_b2 = (const float*)d_in[8];
    const float* peD_W1 = (const float*)d_in[9];
    const float* peD_b1 = (const float*)d_in[10];
    const float* peD_W2 = (const float*)d_in[11];
    const float* peD_b2 = (const float*)d_in[12];

    const float* gP_Wih = (const float*)d_in[13];
    const float* gP_Whh = (const float*)d_in[14];
    const float* gP_bih = (const float*)d_in[15];
    const float* gP_bhh = (const float*)d_in[16];
    const float* gD_Wih = (const float*)d_in[17];
    const float* gD_Whh = (const float*)d_in[18];
    const float* gD_bih = (const float*)d_in[19];
    const float* gD_bhh = (const float*)d_in[20];

    const float* aW[5] = { (const float*)d_in[21], (const float*)d_in[23],
                           (const float*)d_in[25], (const float*)d_in[27],
                           (const float*)d_in[29] };
    const float* ab[5] = { (const float*)d_in[22], (const float*)d_in[24],
                           (const float*)d_in[26], (const float*)d_in[28],
                           (const float*)d_in[30] };

    // workspace carve-out
    char* ws = (char*)d_ws;
    size_t off = 0;
    auto alloc = [&](size_t bytes) -> void* {
        void* p = ws + off;
        off = (off + bytes + 255) & ~(size_t)255;
        return p;
    };
    float*     dist_pts = (float*)alloc((size_t)B_ * T_ * dk * 3 * sizeof(float));
    float*     pseq     = (float*)alloc((size_t)B_ * T_ * ENC_ * sizeof(float));
    float*     dseq     = (float*)alloc((size_t)B_ * T_ * ENC_ * sizeof(float));
    float*     hp0      = (float*)alloc((size_t)B_ * HP_ * sizeof(float));
    float*     hp1      = (float*)alloc((size_t)B_ * HP_ * sizeof(float));
    float*     hd0      = (float*)alloc((size_t)B_ * HD_ * sizeof(float));
    float*     hd1      = (float*)alloc((size_t)B_ * HD_ * sizeof(float));
    _Float16*  lat      = (_Float16*)alloc((size_t)B_ * LAT_PAD_ * sizeof(_Float16));
    _Float16*  act0     = (_Float16*)alloc((size_t)B_ * 1024 * sizeof(_Float16));
    _Float16*  act1     = (_Float16*)alloc((size_t)B_ * 1024 * sizeof(_Float16));
    float* hp[2] = { hp0, hp1 };
    float* hd[2] = { hd0, hd1 };

    // 1) distance bins
    bin_dist_kernel<<<B_ * T_, 256, 0, stream>>>(obs, distIdx, counts, m, dk, dist_pts);

    // 2) encoders (sort skipped: mean-pool is permutation invariant)
    enc_pool_kernel<<<B_ * T_, 256, 0, stream>>>(obs, proxIdx, nullptr, pk,
                                                 peP_W1, peP_b1, peP_W2, peP_b2, pseq);
    enc_pool_kernel<<<B_ * T_, 256, 0, stream>>>(obs, nullptr, dist_pts, dk,
                                                 peD_W1, peD_b1, peD_W2, peD_b2, dseq);

    // 3) GRUs (h0 = 0), ping-pong buffers
    zero_f32_kernel<<<(B_ * HP_ + 255) / 256, 256, 0, stream>>>(hp[0], B_ * HP_);
    zero_f32_kernel<<<(B_ * HD_ + 255) / 256, 256, 0, stream>>>(hd[0], B_ * HD_);
    int cp = 0, cd = 0;
    for (int t = 0; t < T_; ++t) {
        gru_step_kernel<<<B_, 256, 0, stream>>>(pseq, t, gP_Wih, gP_Whh, gP_bih, gP_bhh,
                                                hp[cp], hp[1 - cp], HP_);
        cp ^= 1;
        gru_step_kernel<<<B_, 256, 0, stream>>>(dseq, t, gD_Wih, gD_Whh, gD_bih, gD_bhh,
                                                hd[cd], hd[1 - cd], HD_);
        cd ^= 1;
    }

    // 4) latent
    build_latent_kernel<<<B_, 64, 0, stream>>>(obs, hp[cp], hd[cd], lat);

    // 5) actor MLP: 299->1024->512->256->128->12
    fc_wmma_kernel<<<dim3(B_ / 16, 1024 / 16), 32, 0, stream>>>(lat,  LAT_PAD_, 299,
        aW[0], ab[0], 1024, 1024, 1, act0, nullptr);
    fc_wmma_kernel<<<dim3(B_ / 16, 512 / 16), 32, 0, stream>>>(act0, 1024, 1024,
        aW[1], ab[1], 512, 512, 1, act1, nullptr);
    fc_wmma_kernel<<<dim3(B_ / 16, 256 / 16), 32, 0, stream>>>(act1, 512, 512,
        aW[2], ab[2], 256, 256, 1, act0, nullptr);
    fc_wmma_kernel<<<dim3(B_ / 16, 128 / 16), 32, 0, stream>>>(act0, 256, 256,
        aW[3], ab[3], 128, 128, 1, act1, nullptr);
    fc_wmma_kernel<<<dim3(B_ / 16, 1), 32, 0, stream>>>(act1, 128, 128,
        aW[4], ab[4], 12, 16, 0, nullptr, (float*)d_out);
}
